// HeterogeneousEdgePredictionClassifier_23459111371331
// MI455X (gfx1250) — matrix operand order repaired and checked by reading, not compiled, not run on hardware
//
#include <hip/hip_runtime.h>
#include <hip/hip_bf16.h>
#include <stdint.h>

// ---------------------------------------------------------------------------
// Types for WMMA fragments (CDNA5 wave32)
// ---------------------------------------------------------------------------
typedef __attribute__((ext_vector_type(16))) __bf16 v16bf;
typedef __attribute__((ext_vector_type(8)))  float  v8f;
typedef __attribute__((ext_vector_type(4)))  float  f32x4;
typedef __attribute__((ext_vector_type(4)))  unsigned int u32x4;

union Frag16 { u32x4 q[2]; v16bf v; };

__device__ __forceinline__ unsigned short f2bf(float f) {
  unsigned int u = __float_as_uint(f);
  u += 0x7FFFu + ((u >> 16) & 1u);   // round-to-nearest-even
  return (unsigned short)(u >> 16);
}

// ---------------------------------------------------------------------------
// Prep kernel 1: encoder features f32 -> bf16 (node table becomes 25.6MB, L2-hot)
// ---------------------------------------------------------------------------
__global__ void k_cvt_enc(const float* __restrict__ src,
                          unsigned short* __restrict__ dst, int n) {
  int i = blockIdx.x * blockDim.x + threadIdx.x;
  if (i < n) dst[i] = f2bf(src[i]);
}

// ---------------------------------------------------------------------------
// Prep kernel 2: swizzle W1 (256x128) / W2 (128x64), (in,out) layout, into
// WMMA B-fragment order for v_wmma_f32_16x16x32_bf16:
//   frag (kt,nt): lane -> n = nt*16 + (lane&15), k-half offset = (lane>>4)*8
//   lane element i (0..15) -> k = kt*32 + (i>>3)*16 + half*8 + (i&7)
// stored as ((kt*NT + nt)*32 + lane)*16 + i  => 32B contiguous per lane.
// ---------------------------------------------------------------------------
__global__ void k_swz_w(const float* __restrict__ W1, const float* __restrict__ W2,
                        unsigned short* __restrict__ w1f,
                        unsigned short* __restrict__ w2f) {
  int tid = blockIdx.x * blockDim.x + threadIdx.x;
  if (tid < 32768) {                       // W1: kt 0..7, nt 0..7
    int i = tid & 15, lane = (tid >> 4) & 31, nt = (tid >> 9) & 7, kt = tid >> 12;
    int k = kt * 32 + ((i >> 3) << 4) + ((lane >> 4) << 3) + (i & 7);
    int n = nt * 16 + (lane & 15);
    w1f[tid] = f2bf(W1[k * 128 + n]);
  } else if (tid < 40960) {                // W2: kt 0..3, nt 0..3
    int t2 = tid - 32768;
    int i = t2 & 15, lane = (t2 >> 4) & 31, nt = (t2 >> 9) & 3, kt = t2 >> 11;
    int k = kt * 32 + ((i >> 3) << 4) + ((lane >> 4) << 3) + (i & 7);
    int n = nt * 16 + (lane & 15);
    w2f[t2] = f2bf(W2[k * 64 + n]);
  }
}

// ---------------------------------------------------------------------------
// Fused edge-MLP kernel. 8 wave32 per block, 16 edges per wave.
// LDS: W1 frags 64K | W2 frags 16K | b1 | b2 | W3(transposed) | b3 | 4KB/wave
// ---------------------------------------------------------------------------
#define SM_W1    0
#define SM_W2    65536
#define SM_B1    81920
#define SM_B2    82432
#define SM_W3    82688          // stored transposed: [2][64] f32
#define SM_B3    83200
#define SM_STAGE 83264
#define SM_TOTAL (83264 + 8 * 4096)   // 116032 bytes

__global__ __launch_bounds__(256)
void k_edge_mlp(const unsigned short* __restrict__ encb,
                const unsigned short* __restrict__ w1f,
                const unsigned short* __restrict__ w2f,
                const long long* __restrict__ eidx,
                const float* __restrict__ b1, const float* __restrict__ b2,
                const float* __restrict__ W3, const float* __restrict__ b3,
                float* __restrict__ out, int E, int nTiles) {
  extern __shared__ char smem[];
  unsigned short* sW1 = (unsigned short*)(smem + SM_W1);
  unsigned short* sW2 = (unsigned short*)(smem + SM_W2);
  float* sB1  = (float*)(smem + SM_B1);
  float* sB2  = (float*)(smem + SM_B2);
  float* sW3t = (float*)(smem + SM_W3);
  float* sB3  = (float*)(smem + SM_B3);

  const int tid = threadIdx.x;

  // cooperative stage of weights into LDS (L2-resident global reads)
  {
    const u32x4* g1 = (const u32x4*)w1f;  u32x4* s1 = (u32x4*)sW1;
    #pragma unroll
    for (int i = 0; i < 16; ++i) s1[tid + 256 * i] = g1[tid + 256 * i];   // 4096
    const u32x4* g2 = (const u32x4*)w2f;  u32x4* s2 = (u32x4*)sW2;
    #pragma unroll
    for (int i = 0; i < 4; ++i)  s2[tid + 256 * i] = g2[tid + 256 * i];   // 1024
    if (tid < 128) sB1[tid] = b1[tid];
    if (tid < 64)  sB2[tid] = b2[tid];
    if (tid < 128) {                       // transpose W3 [64][2] -> [2][64]
      int k = tid & 63, c = tid >> 6;
      sW3t[c * 64 + k] = W3[k * 2 + c];
    }
    if (tid < 2)   sB3[tid] = b3[tid];
  }
  __syncthreads();

  const int wave = tid >> 5;
  const int lane = tid & 31;
  const int t = blockIdx.x * 8 + wave;
  if (t >= nTiles) return;                  // wave-uniform: EXEC stays all-ones

  const int m = lane & 15;                  // edge row within tile / matrix row
  const int half = lane >> 4;

  long long e  = (long long)t * 16 + m;
  long long eg = e < (long long)E ? e : (long long)(E - 1);
  const int row = (int)eidx[eg];
  const int col = (int)eidx[(long long)E + eg];
  const unsigned short* rf = encb + (size_t)row * 128;
  const unsigned short* cf = encb + (size_t)col * 128;

  // per-wave LDS staging (reused by layer1 out + layer2 out; DS is in-order)
  unsigned short* h1 = (unsigned short*)(smem + SM_STAGE + wave * 4096); // 16x128 bf16
  float*          h2 = (float*)(smem + SM_STAGE + wave * 4096);          // 16x64  f32

  // ---- layer 1: [16,256] x [256,128], bf16 WMMA --------------------------
  // A fragments: 16-bit 16x32 layout = two contiguous 16B chunks per lane
  Frag16 a1[8];
  #pragma unroll
  for (int kt = 0; kt < 8; ++kt) {
    const unsigned short* src = (kt < 4) ? rf : cf;
    const int kb = (kt & 3) * 32;
    a1[kt].q[0] = *(const u32x4*)(src + kb + half * 8);
    a1[kt].q[1] = *(const u32x4*)(src + kb + 16 + half * 8);
  }
  #pragma unroll
  for (int nt = 0; nt < 8; ++nt) {
    const float bv = sB1[nt * 16 + m];
    v8f acc;
    #pragma unroll
    for (int r = 0; r < 8; ++r) acc[r] = bv;   // C layout: lane=col -> bias bcast
    #pragma unroll
    for (int kt = 0; kt < 8; ++kt) {
      Frag16 bfrag;
      const u32x4* bp = (const u32x4*)(sW1 + ((kt * 8 + nt) * 32 + lane) * 16);
      bfrag.q[0] = bp[0]; bfrag.q[1] = bp[1];
      acc = __builtin_amdgcn_wmma_f32_16x16x32_bf16(
          false, a1[kt].v, false, bfrag.v, (short)0, acc, false, false);
    }
    #pragma unroll
    for (int r = 0; r < 8; ++r) {
      float v = acc[r] > 0.f ? acc[r] : 0.f;   // ReLU
      h1[(r + 8 * half) * 128 + nt * 16 + m] = f2bf(v);
    }
  }

  // ---- layer 2: [16,128] x [128,64] --------------------------------------
  Frag16 a2[4];
  #pragma unroll
  for (int kt = 0; kt < 4; ++kt) {
    const int kb = kt * 32;
    a2[kt].q[0] = *(const u32x4*)(h1 + m * 128 + kb + half * 8);
    a2[kt].q[1] = *(const u32x4*)(h1 + m * 128 + kb + 16 + half * 8);
  }
  #pragma unroll
  for (int nt = 0; nt < 4; ++nt) {
    const float bv = sB2[nt * 16 + m];
    v8f acc;
    #pragma unroll
    for (int r = 0; r < 8; ++r) acc[r] = bv;
    #pragma unroll
    for (int kt = 0; kt < 4; ++kt) {
      Frag16 bfrag;
      const u32x4* bp = (const u32x4*)(sW2 + ((kt * 4 + nt) * 32 + lane) * 16);
      bfrag.q[0] = bp[0]; bfrag.q[1] = bp[1];
      acc = __builtin_amdgcn_wmma_f32_16x16x32_bf16(
          false, a2[kt].v, false, bfrag.v, (short)0, acc, false, false);
    }
    #pragma unroll
    for (int r = 0; r < 8; ++r) {
      float v = acc[r] > 0.f ? acc[r] : 0.f;
      h2[(r + 8 * half) * 64 + nt * 16 + m] = v;  // overwrites h1 (DS in-order)
    }
  }

  // ---- layer 3 + log_softmax: lane = (edge m, class c) -------------------
  // W3 column is contiguous in sW3t -> vector b128 loads for both operands.
  const int c = half;                       // lanes 0-15: class0, 16-31: class1
  const float* hv = h2 + m * 64;
  const float* wv = sW3t + c * 64;
  float dot = sB3[c];
  #pragma unroll
  for (int k = 0; k < 64; k += 4) {
    f32x4 hq = *(const f32x4*)(hv + k);
    f32x4 wq = *(const f32x4*)(wv + k);
    dot = fmaf(hq.x, wq.x, dot);
    dot = fmaf(hq.y, wq.y, dot);
    dot = fmaf(hq.z, wq.z, dot);
    dot = fmaf(hq.w, wq.w, dot);
  }
  const float other = __shfl_xor(dot, 16, 32);          // partner logit (wave32)
  const float mx  = fmaxf(dot, other);
  const float lse = mx + __logf(__expf(dot - mx) + __expf(other - mx));
  if (e < (long long)E) out[e * 2 + c] = dot - lse;
}

// ---------------------------------------------------------------------------
// Launch
// inputs: enc[100000*128] f32, edge_index[2*E] i64, W1[256*128], b1[128],
//         W2[128*64], b2[64], W3[64*2], b3[2]  -> out[E*2] f32 log-softmax
// ---------------------------------------------------------------------------
extern "C" void kernel_launch(void* const* d_in, const int* in_sizes, int n_in,
                              void* d_out, int out_size, void* d_ws, size_t ws_size,
                              hipStream_t stream) {
  const float*     enc  = (const float*)d_in[0];
  const long long* eidx = (const long long*)d_in[1];
  const float*     W1   = (const float*)d_in[2];
  const float*     b1   = (const float*)d_in[3];
  const float*     W2   = (const float*)d_in[4];
  const float*     b2   = (const float*)d_in[5];
  const float*     W3   = (const float*)d_in[6];
  const float*     b3   = (const float*)d_in[7];
  float* out = (float*)d_out;

  const int encN = in_sizes[0];          // 12,800,000
  const int E    = in_sizes[1] / 2;      // 600,000

  // workspace layout: [enc bf16][W1 frags 64KB][W2 frags 16KB]
  size_t off1 = (((size_t)encN * 2) + 255) & ~(size_t)255;
  unsigned short* encb = (unsigned short*)d_ws;
  unsigned short* w1f  = (unsigned short*)((char*)d_ws + off1);
  unsigned short* w2f  = w1f + 32768;

  k_cvt_enc<<<(encN + 255) / 256, 256, 0, stream>>>(enc, encb, encN);
  k_swz_w<<<160, 256, 0, stream>>>(W1, W2, w1f, w2f);

  static_assert(SM_TOTAL == 116032, "lds layout");
  (void)hipFuncSetAttribute((const void*)k_edge_mlp,
                            hipFuncAttributeMaxDynamicSharedMemorySize, SM_TOTAL);

  const int nTiles = (E + 15) / 16;
  const int blocks = (nTiles + 7) / 8;
  k_edge_mlp<<<blocks, 256, SM_TOTAL, stream>>>(encb, w1f, w2f, eidx,
                                                b1, b2, W3, b3, out, E, nTiles);
}
